// NodeMlpType_79568564126388
// MI455X (gfx1250) — matrix operand order repaired and checked by reading, not compiled, not run on hardware
//
#include <hip/hip_runtime.h>

// ---------------------------------------------------------------------------
// MoE-style hard routing on MI455X (gfx1250, wave32, WMMA):
//   out[n] = W2[t]^T relu(W1[t]^T x[n] + b1[t]) + b2[t],  t = node_types[n]
// Nodes are sorted by type into 64-row blocks (type-uniform); each block is a
// 64x128x128 double GEMM: 32 x v_wmma_f32_16x16x32_f16, with all B fragments
// cached in registers and reused across the 4 M-tiles. x rows are gathered
// into LDS with GLOBAL_LOAD_ASYNC_TO_LDS_B128 (ASYNCcnt path).
// ---------------------------------------------------------------------------

typedef _Float16 v16h __attribute__((ext_vector_type(16)));
typedef float    v8f  __attribute__((ext_vector_type(8)));
typedef float    f32x4 __attribute__((ext_vector_type(4)));
typedef unsigned int u32x4 __attribute__((ext_vector_type(4)));
typedef int      v4i  __attribute__((ext_vector_type(4)));

union FragU { v16h v; u32x4 q[2]; };          // 32 bytes: one WMMA A/B operand
union PackU { _Float16 h[2]; unsigned u; };   // half2 <-> dword
union H4U   { _Float16 h[4]; unsigned long long u64; };

#define DIM  128          // DIN == H1 == DOUT == 128
#define XSTR 136          // LDS row stride in halves (pad -> bank spread)
#define TM   64           // rows (nodes) per block

#if __has_builtin(__builtin_amdgcn_global_load_async_to_lds_b128)
#define HAVE_ASYNC_LDS 1
typedef __attribute__((address_space(1))) v4i* gv4i_p;
typedef __attribute__((address_space(3))) v4i* lv4i_p;
#else
#define HAVE_ASYNC_LDS 0
#endif

// --------------------------- routing kernels -------------------------------

__global__ void init_kernel(int* __restrict__ cnt, int* __restrict__ perm,
                            int permLen) {
    int i = blockIdx.x * blockDim.x + threadIdx.x;
    if (i < 64) cnt[i] = 0;                 // cnt[0..31] + cur[32..63]
    if (i < permLen) perm[i] = -1;          // -1 = padding row
}

__global__ void count_kernel(const int* __restrict__ types,
                             int* __restrict__ cnt, int N) {
    int i = blockIdx.x * blockDim.x + threadIdx.x;
    if (i < N) atomicAdd(&cnt[types[i] & 31], 1);
}

__global__ void prefix_kernel(const int* __restrict__ cnt, int* __restrict__ cur,
                              int* __restrict__ po, int* __restrict__ nBlocks,
                              int* __restrict__ blkType, int T) {
    if (threadIdx.x != 0 || blockIdx.x != 0) return;
    int pos = 0;                            // in units of 64-row blocks
    for (int t = 0; t < T; ++t) {
        po[t] = pos * TM;                   // 64-aligned segment start (rows)
        cur[t] = 0;
        int nb = (cnt[t] + TM - 1) / TM;
        for (int i = 0; i < nb; ++i) blkType[pos + i] = t;
        pos += nb;
    }
    *nBlocks = pos;
}

__global__ void scatter_kernel(const int* __restrict__ types,
                               int* __restrict__ cur, const int* __restrict__ po,
                               int* __restrict__ perm, int N) {
    int i = blockIdx.x * blockDim.x + threadIdx.x;
    if (i < N) {
        int t = types[i] & 31;
        int r = atomicAdd(&cur[t], 1);
        perm[po[t] + r] = i;
    }
}

// ------------------- weight f32 -> f16 fragment swizzle --------------------
// Fragment f = (t*4 + kTile)*8 + nTile holds a 32x16 f16 B operand as 256
// dwords: dword d = c*128 + lane*4 + q, so each lane's 8 dwords are two
// contiguous b128 loads. Contiguous-k B layout used here (A matches): lanes 0-15 hold K=0..15
// (2/dword), lanes 16-31 K=16..31, column N = lane%16, low half = even K.

__global__ void swizzle_kernel(const float* __restrict__ W1,
                               const float* __restrict__ W2,
                               unsigned* __restrict__ w1s,
                               unsigned* __restrict__ w2s, int T) {
    int i = blockIdx.x * blockDim.x + threadIdx.x;
    int total = T * 8192;                   // dwords per matrix stack
    if (i >= 2 * total) return;
    const float* W  = (i < total) ? W1  : W2;
    unsigned*    Wd = (i < total) ? w1s : w2s;
    int d    = (i < total) ? i : i - total;
    int f    = d >> 8;
    int r    = d & 255;
    int c    = r >> 7;
    int lane = (r >> 2) & 31;
    int q    = r & 3;
    int j    = c * 4 + q;                   // dword id 0..7 within lane
    int t    = f >> 5;
    int rem  = f & 31;
    int kt   = rem >> 3;
    int nt   = rem & 7;
    int n    = nt * 16 + (lane & 15);
    int k0   = kt * 32 + ((lane >> 4) << 4) + 2 * j;
    PackU p;
    p.h[0] = (_Float16)W[t * 16384 + k0 * 128 + n];
    p.h[1] = (_Float16)W[t * 16384 + (k0 + 1) * 128 + n];
    Wd[d] = p.u;
}

// ------------------------------ GEMM kernel --------------------------------

__global__ __launch_bounds__(256)
void gemm_kernel(const float* __restrict__ x,
                 const unsigned* __restrict__ w1s,
                 const unsigned* __restrict__ w2s,
                 const float* __restrict__ b1,
                 const float* __restrict__ b2,
                 const int* __restrict__ perm,
                 const int* __restrict__ blkType,
                 const int* __restrict__ nBlocksPtr,
                 float* __restrict__ out) {
    // lds_stage (f32 x tile, 32 KB) is dead after the f16 convert; lds_h
    // (64*XSTR halves = 17.4 KB) overlays it across a barrier.
    __shared__ __align__(16) float    lds_stage[TM * DIM];
    __shared__ __align__(16) _Float16 lds_x[TM * XSTR];
    __shared__ float lds_b1[DIM];
    __shared__ float lds_b2[DIM];
    __shared__ int   lds_perm[TM];
    _Float16* lds_h = (_Float16*)lds_stage;

    int blk = blockIdx.x;
    if (blk >= *nBlocksPtr) return;         // block-uniform exit
    int t    = blkType[blk];
    int tid  = threadIdx.x;
    int wave = tid >> 5;                    // 0..7
    int lane = tid & 31;

    if (tid < TM)  lds_perm[tid] = perm[blk * TM + tid];
    if (tid < 128) lds_b1[tid] = b1[t * DIM + tid];
    else           lds_b2[tid - 128] = b2[t * DIM + (tid - 128)];

#if HAVE_ASYNC_LDS
    // Async gather: 64 rows x 512B. Per j, all 32 lanes of a wave copy one
    // row (wave-uniform perm lookup -> scalar load), 16B per lane, straight
    // into LDS via the ASYNCcnt path (no VGPR round-trip).
    {
        char* stage_c = (char*)lds_stage;
        #pragma unroll
        for (int j = 0; j < 8; ++j) {
            int row  = wave + 8 * j;        // 0..63, uniform across the wave
            int node = perm[blk * TM + row];
            const float* gsrc = x + (size_t)(node < 0 ? 0 : node) * DIM + lane * 4;
            __builtin_amdgcn_global_load_async_to_lds_b128(
                (gv4i_p)(void*)gsrc,
                (lv4i_p)(void*)(stage_c + (row * DIM + lane * 4) * 4),
                0, 0);
        }
#if __has_builtin(__builtin_amdgcn_s_wait_asynccnt)
        __builtin_amdgcn_s_wait_asynccnt(0);
#else
        asm volatile("s_wait_asynccnt 0x0" ::: "memory");
#endif
    }
    __syncthreads();                        // all waves' async copies + perm/bias
    {
        int r  = tid >> 2;                  // row 0..63
        int c0 = (tid & 3) * 32;            // 32 cols per thread
        int node = lds_perm[r];
        #pragma unroll
        for (int j = 0; j < 8; ++j) {       // b128 load -> pack -> b64 store
            f32x4 v4 = *(const f32x4*)(lds_stage + r * DIM + c0 + j * 4);
            H4U o;
            #pragma unroll
            for (int e = 0; e < 4; ++e)
                o.h[e] = (_Float16)(node >= 0 ? v4[e] : 0.0f);
            *(unsigned long long*)(lds_x + r * XSTR + c0 + j * 4) = o.u64;
        }
    }
#else
    // Fallback: direct gather + convert (vectorized).
    {
        int r  = tid >> 2;
        int c0 = (tid & 3) * 32;
        int node = perm[blk * TM + r];
        const float* src = x + (size_t)(node < 0 ? 0 : node) * DIM + c0;
        #pragma unroll
        for (int j = 0; j < 8; ++j) {
            f32x4 v4 = *(const f32x4*)(src + j * 4);
            H4U o;
            #pragma unroll
            for (int e = 0; e < 4; ++e)
                o.h[e] = (_Float16)(node >= 0 ? v4[e] : 0.0f);
            *(unsigned long long*)(lds_x + r * XSTR + c0 + j * 4) = o.u64;
        }
    }
#endif
    __syncthreads();                        // stage consumed; lds_h may be written

    int lm   = lane & 15;                   // A row / B-C column index
    int sel  = lane >> 4;                   // lane half select
    int ncol = (wave << 4) + lm;

    // Cache ALL B fragments (both layers) in registers: 4 kt x 8 dwords x 2.
    FragU B1[4], B2[4];
    #pragma unroll
    for (int kt = 0; kt < 4; ++kt) {
        const unsigned* fb1 = w1s + (((t * 4 + kt) * 8 + wave) << 8);
        B1[kt].q[0] = *(const u32x4*)(fb1 + lane * 4);
        B1[kt].q[1] = *(const u32x4*)(fb1 + 128 + lane * 4);
        const unsigned* fb2 = w2s + (((t * 4 + kt) * 8 + wave) << 8);
        B2[kt].q[0] = *(const u32x4*)(fb2 + lane * 4);
        B2[kt].q[1] = *(const u32x4*)(fb2 + 128 + lane * 4);
    }

    // ----- layer 1: h = relu(x*W1 + b1); 4 M-tiles x 4 K-tiles -----
    float bias1 = lds_b1[ncol];
    #pragma unroll
    for (int mt = 0; mt < 4; ++mt) {
        v8f acc = {0.f, 0.f, 0.f, 0.f, 0.f, 0.f, 0.f, 0.f};
        const unsigned* ax = (const unsigned*)lds_x + (mt * 16 + lm) * (XSTR / 2);
        #pragma unroll
        for (int kt = 0; kt < 4; ++kt) {
            FragU A;   // lanes 0-15: v0..7=K0..15; lanes 16-31: K16..31
            A.q[0] = *(const u32x4*)(ax + kt * 16 + sel * 8);
            A.q[1] = *(const u32x4*)(ax + kt * 16 + 4 + sel * 8);
            acc = __builtin_amdgcn_wmma_f32_16x16x32_f16(
                false, A.v, false, B1[kt].v, (short)0, acc, false, false);
        }
        #pragma unroll
        for (int v = 0; v < 8; ++v) {       // C/D layout: vgpr v -> M = v + 8*sel
            float h = acc[v] + bias1;
            h = h > 0.0f ? h : 0.0f;
            lds_h[(mt * 16 + v + sel * 8) * XSTR + ncol] = (_Float16)h;
        }
    }
    __syncthreads();

    // ----- layer 2: y = h*W2 + b2, scatter rows back -----
    float bias2 = lds_b2[ncol];
    #pragma unroll
    for (int mt = 0; mt < 4; ++mt) {
        v8f acc = {0.f, 0.f, 0.f, 0.f, 0.f, 0.f, 0.f, 0.f};
        const unsigned* ah = (const unsigned*)lds_h + (mt * 16 + lm) * (XSTR / 2);
        #pragma unroll
        for (int kt = 0; kt < 4; ++kt) {
            FragU A;
            A.q[0] = *(const u32x4*)(ah + kt * 16 + sel * 8);
            A.q[1] = *(const u32x4*)(ah + kt * 16 + 4 + sel * 8);
            acc = __builtin_amdgcn_wmma_f32_16x16x32_f16(
                false, A.v, false, B2[kt].v, (short)0, acc, false, false);
        }
        #pragma unroll
        for (int v = 0; v < 8; ++v) {
            int mrow = mt * 16 + v + sel * 8;
            int node = lds_perm[mrow];
            if (node >= 0) out[(size_t)node * DIM + ncol] = acc[v] + bias2;
        }
    }
}

// ------------------------------- launcher ----------------------------------

extern "C" void kernel_launch(void* const* d_in, const int* in_sizes, int n_in,
                              void* d_out, int out_size, void* d_ws, size_t ws_size,
                              hipStream_t stream) {
    const float* x     = (const float*)d_in[0];
    const float* W1    = (const float*)d_in[1];
    const float* b1    = (const float*)d_in[2];
    const float* W2    = (const float*)d_in[3];
    const float* b2    = (const float*)d_in[4];
    const int*   types = (const int*)d_in[5];
    float* out = (float*)d_out;

    int N = in_sizes[0] / DIM;                  // 32768
    int T = in_sizes[1] / (DIM * DIM);          // 17
    int maxBlocks = (N + TM - 1) / TM + T;      // padded 64-row block bound

    // Workspace layout
    int* cnt     = (int*)d_ws;                  // [32] counts
    int* cur     = cnt + 32;                    // [32] scatter cursors
    int* po      = cnt + 64;                    // [32] padded segment offsets
    int* nBlocks = cnt + 96;                    // [1]
    int* blkType = cnt + 256;                   // [maxBlocks]
    int* perm    = blkType + maxBlocks;         // [maxBlocks*TM]
    size_t ofs = (size_t)(256 + maxBlocks + (size_t)maxBlocks * TM) * sizeof(int);
    ofs = (ofs + 255) & ~(size_t)255;
    unsigned* w1s = (unsigned*)((char*)d_ws + ofs);      // T*8192 dwords
    unsigned* w2s = w1s + (size_t)T * 8192;              // T*8192 dwords

    int permLen = maxBlocks * TM;
    init_kernel<<<(permLen + 255) / 256, 256, 0, stream>>>(cnt, perm, permLen);
    count_kernel<<<(N + 255) / 256, 256, 0, stream>>>(types, cnt, N);
    prefix_kernel<<<1, 64, 0, stream>>>(cnt, cur, po, nBlocks, blkType, T);
    scatter_kernel<<<(N + 255) / 256, 256, 0, stream>>>(types, cur, po, perm, N);
    swizzle_kernel<<<(2 * T * 8192 + 255) / 256, 256, 0, stream>>>(W1, W2, w1s, w2s, T);
    gemm_kernel<<<maxBlocks, 256, 0, stream>>>(x, w1s, w2s, b1, b2,
                                               perm, blkType, nBlocks, out);
}
